// GenerativeRetrievalModel_4776003633237
// MI455X (gfx1250) — matrix operand order repaired and checked by reading, not compiled, not run on hardware
//
#include <hip/hip_runtime.h>
#include <hip/hip_bf16.h>
#include <math.h>

typedef unsigned short u16;
typedef __attribute__((ext_vector_type(8)))  __bf16 v8bf;
typedef __attribute__((ext_vector_type(16))) __bf16 v16bf;
typedef __attribute__((ext_vector_type(8)))  float  v8f;
typedef int v4i __attribute__((vector_size(4 * sizeof(int))));

#define NEG_INF_F (-1.0e9f)

// LDS tile row stride (elements): 64 data + 8 pad -> 36-dword row pitch,
// which maps 16 consecutive rows onto 16 distinct banks (36n mod 64 perm).
#define LDS_PITCH 72

// ---------------------------------------------------------------- helpers
__device__ __forceinline__ u16 f32_to_bf16_bits(float f) {
  union { float f; unsigned u; } x; x.f = f;
  unsigned r = x.u + 0x7FFFu + ((x.u >> 16) & 1u);   // round-nearest-even
  return (u16)(r >> 16);
}

__device__ __forceinline__ v8f zero_v8f() {
  v8f z;
#pragma unroll
  for (int i = 0; i < 8; ++i) z[i] = 0.0f;
  return z;
}

// A fragment (16x32 bf16, MxK): lane holds row m=lane%16, 16 consecutive
// K at (lane/16)*16.
__device__ __forceinline__ v16bf load_frag_a(const u16* p0) {
  v8bf lo = *(const v8bf*)p0;
  v8bf hi = *(const v8bf*)(p0 + 8);
  return __builtin_shufflevector(lo, hi, 0,1,2,3,4,5,6,7,8,9,10,11,12,13,14,15);
}
// B fragment (32x16 bf16, KxN) from an N-major (N x K) array: lane holds
// col n=lane%16, 16 consecutive K at (lane/16)*16.
__device__ __forceinline__ v16bf load_frag_b(const u16* p0) {
  v8bf lo = *(const v8bf*)p0;
  v8bf hi = *(const v8bf*)(p0 + 8);
  return __builtin_shufflevector(lo, hi, 0,1,2,3,4,5,6,7,8,9,10,11,12,13,14,15);
}

// ---------------------------------------------------------------- async copy
// 16B global -> LDS async copy (ASYNCcnt-tracked CDNA5 path).
__device__ __forceinline__ void async_copy_b128(const u16* g, u16* l) {
#if defined(__has_builtin) && __has_builtin(__builtin_amdgcn_global_load_async_to_lds_b128)
  __builtin_amdgcn_global_load_async_to_lds_b128(
      (__attribute__((address_space(1))) v4i*)(unsigned long long)(const void*)g,
      (__attribute__((address_space(3))) v4i*)(unsigned long long)(void*)l,
      0, 0);
#else
  unsigned lds_addr = (unsigned)(unsigned long long)(void*)l;
  asm volatile("global_load_async_to_lds_b128 %0, %1, off"
               :: "v"(lds_addr), "v"(g) : "memory");
#endif
}

__device__ __forceinline__ void wait_async_le6() {
#if defined(__has_builtin) && __has_builtin(__builtin_amdgcn_s_wait_asynccnt)
  __builtin_amdgcn_s_wait_asynccnt(6);
#else
  asm volatile("s_wait_asynccnt 0x6" ::: "memory");
#endif
}
__device__ __forceinline__ void wait_async_le0() {
#if defined(__has_builtin) && __has_builtin(__builtin_amdgcn_s_wait_asynccnt)
  __builtin_amdgcn_s_wait_asynccnt(0);
#else
  asm volatile("s_wait_asynccnt 0x0" ::: "memory");
#endif
}

// ---------------------------------------------------------------- GEMM
// C[MxN] = A[MxK] * B^T[NxK]  (bf16 in, f32 accumulate), batched over grid.z.
// Workgroup tile 64(M) x 128(N); 8 waves; wave tile 16(M) x 64(N).
// K staged in 64-wide slices through double-buffered LDS with async copies
// (6 b128 copies per thread per slice); 8 WMMAs per barrier pair per wave.
enum { EPI_BF16 = 0, EPI_BF16_TRANS, EPI_MASK, EPI_RES_F32, EPI_RES_BF16, EPI_BIAS_F32 };

// Stage a 64x64 A tile (512 chunks) + 128x64 B tile (1024 chunks) of 16B.
__device__ __forceinline__ void stage_tile64(const u16* Ag, long lda,
                                             const u16* Bg, long ldb,
                                             u16* sa, u16* sb, int tid) {
#pragma unroll
  for (int i = 0; i < 2; ++i) {
    int c = tid + i * 256;
    async_copy_b128(Ag + (long)(c >> 3) * lda + (c & 7) * 8,
                    sa + (c >> 3) * LDS_PITCH + (c & 7) * 8);
  }
#pragma unroll
  for (int i = 0; i < 4; ++i) {
    int c = tid + i * 256;
    async_copy_b128(Bg + (long)(c >> 3) * ldb + (c & 7) * 8,
                    sb + (c >> 3) * LDS_PITCH + (c & 7) * 8);
  }
}

__device__ __forceinline__ void consume_slice64(const u16* la, const u16* lb,
                                                int lane16, int halfw,
                                                v8f& acc0, v8f& acc1,
                                                v8f& acc2, v8f& acc3) {
#pragma unroll
  for (int ks = 0; ks < 64; ks += 32) {
    v16bf a  = load_frag_a(la + lane16 * LDS_PITCH + ks + halfw * 16);
    v16bf b0 = load_frag_b(lb + (lane16 +  0) * LDS_PITCH + ks + halfw * 16);
    v16bf b1 = load_frag_b(lb + (lane16 + 16) * LDS_PITCH + ks + halfw * 16);
    v16bf b2 = load_frag_b(lb + (lane16 + 32) * LDS_PITCH + ks + halfw * 16);
    v16bf b3 = load_frag_b(lb + (lane16 + 48) * LDS_PITCH + ks + halfw * 16);
    acc0 = __builtin_amdgcn_wmma_f32_16x16x32_bf16(false, a, false, b0, (short)0, acc0, false, false);
    acc1 = __builtin_amdgcn_wmma_f32_16x16x32_bf16(false, a, false, b1, (short)0, acc1, false, false);
    acc2 = __builtin_amdgcn_wmma_f32_16x16x32_bf16(false, a, false, b2, (short)0, acc2, false, false);
    acc3 = __builtin_amdgcn_wmma_f32_16x16x32_bf16(false, a, false, b3, (short)0, acc3, false, false);
  }
}

template <int EPI>
__global__ __launch_bounds__(256) void gemm_bf16_wmma(
    const u16* __restrict__ A, const u16* __restrict__ Bm,
    long sA, long sB, int lda, int ldb, int Kdim,
    float* __restrict__ Cf, u16* __restrict__ Cb, long sC, int ldc,
    const float* __restrict__ Res, long sRes, int ldres,
    const float* __restrict__ bias, float scale)
{
  __shared__ __align__(16) u16 sAb[2][64 * LDS_PITCH];    // 2 x 9 KB
  __shared__ __align__(16) u16 sBb[2][128 * LDS_PITCH];   // 2 x 18 KB

  const int tid    = threadIdx.x;
  const int wave   = tid >> 5;
  const int lane   = tid & 31;
  const int lane16 = lane & 15;
  const int halfw  = lane >> 4;
  const int z      = blockIdx.z;

  const int m0 = blockIdx.y * 64  + (wave & 3) * 16;
  const int n0 = blockIdx.x * 128 + (wave >> 2) * 64;

  if (EPI == EPI_MASK) {
    // causal early-out: whole block tile strictly above the diagonal
    if ((int)blockIdx.x * 128 > (int)blockIdx.y * 64 + 63) {
      float* Crow = Cf + z * sC;
#pragma unroll
      for (int j = 0; j < 4; ++j)
#pragma unroll
        for (int r = 0; r < 8; ++r) {
          int m = m0 + r + 8 * halfw;
          int n = n0 + j * 16 + lane16;
          Crow[(long)m * ldc + n] = NEG_INF_F;
        }
      return;
    }
  }

  const u16* Ag = A  + z * sA + (long)(blockIdx.y * 64)  * lda;
  const u16* Bg = Bm + z * sB + (long)(blockIdx.x * 128) * ldb;
  const u16* la0 = &sAb[0][(wave & 3) * (16 * LDS_PITCH)];
  const u16* la1 = &sAb[1][(wave & 3) * (16 * LDS_PITCH)];
  const u16* lb0 = &sBb[0][(wave >> 2) * (64 * LDS_PITCH)];
  const u16* lb1 = &sBb[1][(wave >> 2) * (64 * LDS_PITCH)];

  v8f acc0 = zero_v8f(), acc1 = zero_v8f(), acc2 = zero_v8f(), acc3 = zero_v8f();

  const int KT = Kdim >> 6;                 // 64-wide K slices (KT >= 8 here)
  stage_tile64(Ag, lda, Bg, ldb, sAb[0], sBb[0], tid);

  // steady state: branch-free stage-ahead + consume
  for (int kt = 0; kt < KT - 1; ++kt) {
    stage_tile64(Ag + (kt + 1) * 64, lda, Bg + (kt + 1) * 64, ldb,
                 sAb[(kt + 1) & 1], sBb[(kt + 1) & 1], tid);
    wait_async_le6();                       // slice kt landed (in-order retire)
    __syncthreads();
    consume_slice64((kt & 1) ? la1 : la0, (kt & 1) ? lb1 : lb0,
                    lane16, halfw, acc0, acc1, acc2, acc3);
    __syncthreads();                        // safe to restage this buffer
  }
  // epilogue slice
  wait_async_le0();
  __syncthreads();
  consume_slice64(((KT - 1) & 1) ? la1 : la0, ((KT - 1) & 1) ? lb1 : lb0,
                  lane16, halfw, acc0, acc1, acc2, acc3);

  v8f accs[4] = {acc0, acc1, acc2, acc3};
#pragma unroll
  for (int j = 0; j < 4; ++j) {
#pragma unroll
    for (int r = 0; r < 8; ++r) {
      int m = m0 + r + 8 * halfw;          // C layout: VGPR r -> M=r / M=8+r
      int n = n0 + j * 16 + lane16;        // N striped across lanes
      float v = accs[j][r];
      if (EPI == EPI_BF16) {
        Cb[z * sC + (long)m * ldc + n] = f32_to_bf16_bits(v);
      } else if (EPI == EPI_BF16_TRANS) {
        Cb[z * sC + (long)n * ldc + m] = f32_to_bf16_bits(v);
      } else if (EPI == EPI_MASK) {
        Cf[z * sC + (long)m * ldc + n] = (n <= m) ? v * scale : NEG_INF_F;
      } else if (EPI == EPI_RES_F32) {
        Cf[z * sC + (long)m * ldc + n] = Res[z * sRes + (long)m * ldres + n] + v;
      } else if (EPI == EPI_RES_BF16) {
        Cb[z * sC + (long)m * ldc + n] =
            f32_to_bf16_bits(Res[z * sRes + (long)m * ldres + n] + v);
      } else { // EPI_BIAS_F32
        Cf[z * sC + (long)m * ldc + n] = v + bias[n];
      }
    }
  }
}

// ---------------------------------------------------------------- misc kernels
__global__ void k_f32_to_bf16(const float* __restrict__ in, u16* __restrict__ out, long n) {
  long i = (long)blockIdx.x * blockDim.x + threadIdx.x;
  long stride = (long)gridDim.x * blockDim.x;
  for (; i < n; i += stride) out[i] = f32_to_bf16_bits(in[i]);
}

// W (Kd x Nd) f32 row-major -> Wt (Nd x Kd) bf16 row-major
__global__ __launch_bounds__(256) void k_transpose_bf16(
    const float* __restrict__ W, u16* __restrict__ Wt, int Kd, int Nd) {
  __shared__ float tile[32][33];
  int n0 = blockIdx.x * 32, k0 = blockIdx.y * 32;
  int tx = threadIdx.x, ty = threadIdx.y;
  for (int i = ty; i < 32; i += 8)
    tile[i][tx] = W[(long)(k0 + i) * Nd + n0 + tx];
  __syncthreads();
  for (int i = ty; i < 32; i += 8)
    Wt[(long)(n0 + i) * Kd + k0 + tx] = f32_to_bf16_bits(tile[tx][i]);
}

// x2[b, r, :] = r<S ? h1[b,r,:] : (r<S+T ? tok_emb[tok[b,r-S],:] : 0)
__global__ __launch_bounds__(256) void k_build_x2(
    const float* __restrict__ h1f, const float* __restrict__ tok_emb,
    const int* __restrict__ toks, float* __restrict__ x2f, u16* __restrict__ x2b,
    int S, int T, int Lp, int D) {
  int row = blockIdx.x;
  int b = row / Lp, r = row - b * Lp;
  long dst = (long)row * D;
  for (int d = threadIdx.x; d < D; d += blockDim.x) {
    float v;
    if (r < S)          v = h1f[((long)b * S + r) * D + d];
    else if (r < S + T) v = tok_emb[(long)toks[b * T + (r - S)] * D + d];
    else                v = 0.0f;
    x2f[dst + d] = v;
    x2b[dst + d] = f32_to_bf16_bits(v);
  }
}

__global__ __launch_bounds__(256) void k_softmax_row(
    const float* __restrict__ S, u16* __restrict__ P, int Lp) {
  __shared__ float red[256];
  long row = blockIdx.x;
  const float* s = S + row * Lp;
  u16* p = P + row * Lp;
  int tid = threadIdx.x;
  float mx = -3.4e38f;
  for (int i = tid; i < Lp; i += 256) mx = fmaxf(mx, s[i]);
  red[tid] = mx; __syncthreads();
  for (int off = 128; off > 0; off >>= 1) {
    if (tid < off) red[tid] = fmaxf(red[tid], red[tid + off]);
    __syncthreads();
  }
  mx = red[0]; __syncthreads();
  float sum = 0.f;
  for (int i = tid; i < Lp; i += 256) sum += __expf(s[i] - mx);
  red[tid] = sum; __syncthreads();
  for (int off = 128; off > 0; off >>= 1) {
    if (tid < off) red[tid] += red[tid + off];
    __syncthreads();
  }
  float inv = 1.0f / red[0];
  for (int i = tid; i < Lp; i += 256) p[i] = f32_to_bf16_bits(__expf(s[i] - mx) * inv);
}

__global__ void k_gather_ctx(const u16* __restrict__ h2b, u16* __restrict__ ctx,
                             int Lp, int D, int S, int T) {
  int row = blockIdx.x;                 // b*T + t
  int b = row / T, t = row - b * T;
  const u16* src = h2b + ((long)b * Lp + (S - 1 + t)) * D;
  u16* dst = ctx + (long)row * D;
  for (int d = threadIdx.x; d < D; d += blockDim.x) dst[d] = src[d];
}

__global__ __launch_bounds__(256) void k_loss(
    const float* __restrict__ logits, const int* __restrict__ toks,
    float* __restrict__ out, int V, float invB) {
  __shared__ float red[256];
  int row = blockIdx.x;
  const float* l = logits + (long)row * V;
  int tid = threadIdx.x;
  float mx = -3.4e38f;
  for (int i = tid; i < V; i += 256) mx = fmaxf(mx, l[i]);
  red[tid] = mx; __syncthreads();
  for (int off = 128; off > 0; off >>= 1) {
    if (tid < off) red[tid] = fmaxf(red[tid], red[tid + off]);
    __syncthreads();
  }
  mx = red[0]; __syncthreads();
  float sum = 0.f;
  for (int i = tid; i < V; i += 256) sum += __expf(l[i] - mx);
  red[tid] = sum; __syncthreads();
  for (int off = 128; off > 0; off >>= 1) {
    if (tid < off) red[tid] += red[tid + off];
    __syncthreads();
  }
  if (tid == 0) {
    float nll = logf(red[0]) + mx - l[toks[row]];
    atomicAdd(out, nll * invB);
  }
}

__global__ void k_zero(float* p, int n) {
  int i = blockIdx.x * blockDim.x + threadIdx.x;
  if (i < n) p[i] = 0.f;
}

// ---------------------------------------------------------------- host side
static const int cB = 64, cS = 512, cT = 8, cD = 1024, cV = 32768;
static const int LP1 = 512, LP2 = 640;     // pass-2 L=520 padded to 640 (=128*5)

static void run_attn_pass(const u16* xbf, const float* resf, int Lp,
                          const u16* WqT, const u16* WkT, const u16* WvT, const u16* WoT,
                          u16* Q, u16* Kb, u16* Vt, float* Sc, u16* P, u16* attn,
                          float* houtf, u16* houtbf, hipStream_t stream) {
  dim3 blk(256);
  const long sX = (long)Lp * cD;
  // Q = X Wq^T ; K = X Wk^T (bf16, row-major L x D)
  gemm_bf16_wmma<EPI_BF16><<<dim3(cD / 128, Lp / 64, cB), blk, 0, stream>>>(
      xbf, WqT, sX, 0, cD, cD, cD, nullptr, Q, sX, cD, nullptr, 0, 0, nullptr, 0.f);
  gemm_bf16_wmma<EPI_BF16><<<dim3(cD / 128, Lp / 64, cB), blk, 0, stream>>>(
      xbf, WkT, sX, 0, cD, cD, cD, nullptr, Kb, sX, cD, nullptr, 0, 0, nullptr, 0.f);
  // V stored transposed (D x L) so it is the B-operand of A*V directly
  gemm_bf16_wmma<EPI_BF16_TRANS><<<dim3(cD / 128, Lp / 64, cB), blk, 0, stream>>>(
      xbf, WvT, sX, 0, cD, cD, cD, nullptr, Vt, (long)cD * Lp, Lp, nullptr, 0, 0, nullptr, 0.f);
  // scores = mask(QK^T / sqrt(D))
  gemm_bf16_wmma<EPI_MASK><<<dim3(Lp / 128, Lp / 64, cB), blk, 0, stream>>>(
      Q, Kb, sX, sX, cD, cD, cD, Sc, nullptr, (long)Lp * Lp, Lp, nullptr, 0, 0,
      nullptr, 0.03125f);
  // row softmax -> bf16 probs
  k_softmax_row<<<cB * Lp, 256, 0, stream>>>(Sc, P, Lp);
  // attn = P * V
  gemm_bf16_wmma<EPI_BF16><<<dim3(cD / 128, Lp / 64, cB), blk, 0, stream>>>(
      P, Vt, (long)Lp * Lp, (long)cD * Lp, Lp, Lp, Lp, nullptr, attn, sX, cD,
      nullptr, 0, 0, nullptr, 0.f);
  // h = x + attn * Wo^T
  if (houtf)
    gemm_bf16_wmma<EPI_RES_F32><<<dim3(cD / 128, Lp / 64, cB), blk, 0, stream>>>(
        attn, WoT, sX, 0, cD, cD, cD, houtf, nullptr, sX, cD, resf, sX, cD, nullptr, 0.f);
  else
    gemm_bf16_wmma<EPI_RES_BF16><<<dim3(cD / 128, Lp / 64, cB), blk, 0, stream>>>(
        attn, WoT, sX, 0, cD, cD, cD, nullptr, houtbf, sX, cD, resf, sX, cD, nullptr, 0.f);
}

extern "C" void kernel_launch(void* const* d_in, const int* in_sizes, int n_in,
                              void* d_out, int out_size, void* d_ws, size_t ws_size,
                              hipStream_t stream) {
  const float* x1      = (const float*)d_in[0];   // (B,S,D)
  const int*   toks    = (const int*)  d_in[3];   // (B,T)
  const float* Wq      = (const float*)d_in[5];
  const float* Wk      = (const float*)d_in[6];
  const float* Wv      = (const float*)d_in[7];
  const float* Wo      = (const float*)d_in[8];
  const float* tok_emb = (const float*)d_in[9];
  const float* head_w  = (const float*)d_in[10];  // (D,V)
  const float* head_b  = (const float*)d_in[11];
  float* out = (float*)d_out;

  // ---- workspace bump allocator (256B aligned)
  size_t off = 0;
  auto alloc = [&](size_t bytes) -> char* {
    char* p = (char*)d_ws + off;
    off = (off + bytes + 255) & ~(size_t)255;
    return p;
  };
  u16*   WqT    = (u16*)  alloc((size_t)cD * cD * 2);
  u16*   WkT    = (u16*)  alloc((size_t)cD * cD * 2);
  u16*   WvT    = (u16*)  alloc((size_t)cD * cD * 2);
  u16*   WoT    = (u16*)  alloc((size_t)cD * cD * 2);
  u16*   headWT = (u16*)  alloc((size_t)cV * cD * 2);
  u16*   x1bf   = (u16*)  alloc((size_t)cB * LP1 * cD * 2);
  u16*   Q      = (u16*)  alloc((size_t)cB * LP2 * cD * 2);
  u16*   Kb     = (u16*)  alloc((size_t)cB * LP2 * cD * 2);
  u16*   Vt     = (u16*)  alloc((size_t)cB * cD * LP2 * 2);
  float* Sc     = (float*)alloc((size_t)cB * LP2 * LP2 * 4);
  u16*   P      = (u16*)  alloc((size_t)cB * LP2 * LP2 * 2);
  u16*   attn   = (u16*)  alloc((size_t)cB * LP2 * cD * 2);
  float* h1f    = (float*)alloc((size_t)cB * LP1 * cD * 4);
  float* x2f    = (float*)alloc((size_t)cB * LP2 * cD * 4);
  u16*   x2bf   = (u16*)  alloc((size_t)cB * LP2 * cD * 2);
  u16*   h2bf   = (u16*)  alloc((size_t)cB * LP2 * cD * 2);
  u16*   ctxbf  = (u16*)  alloc((size_t)cB * cT * cD * 2);
  float* logits = (float*)alloc((size_t)cB * cT * cV * 4);
  (void)ws_size; (void)in_sizes; (void)n_in; (void)out_size;

  // ---- prep: convert / transpose weights + input
  k_f32_to_bf16<<<4096, 256, 0, stream>>>(x1, x1bf, (long)cB * LP1 * cD);
  dim3 tblk(32, 8);
  k_transpose_bf16<<<dim3(cD / 32, cD / 32), tblk, 0, stream>>>(Wq, WqT, cD, cD);
  k_transpose_bf16<<<dim3(cD / 32, cD / 32), tblk, 0, stream>>>(Wk, WkT, cD, cD);
  k_transpose_bf16<<<dim3(cD / 32, cD / 32), tblk, 0, stream>>>(Wv, WvT, cD, cD);
  k_transpose_bf16<<<dim3(cD / 32, cD / 32), tblk, 0, stream>>>(Wo, WoT, cD, cD);
  k_transpose_bf16<<<dim3(cV / 32, cD / 32), tblk, 0, stream>>>(head_w, headWT, cD, cV);

  // ---- pass 1: causal attention over history (L=512)
  run_attn_pass(x1bf, x1, LP1, WqT, WkT, WvT, WoT, Q, Kb, Vt, Sc, P, attn,
                h1f, nullptr, stream);

  // ---- build padded pass-2 input: [h1 | tok_emb[toks] | 0]
  k_build_x2<<<cB * LP2, 256, 0, stream>>>(h1f, tok_emb, toks, x2f, x2bf,
                                           cS, cT, LP2, cD);

  // ---- pass 2: causal attention over full sequence (L=520 padded to 640)
  run_attn_pass(x2bf, x2f, LP2, WqT, WkT, WvT, WoT, Q, Kb, Vt, Sc, P, attn,
                nullptr, h2bf, stream);

  // ---- ctx rows S-1 .. S+T-2 -> head GEMM -> NLL
  k_gather_ctx<<<cB * cT, 256, 0, stream>>>(h2bf, ctxbf, LP2, cD, cS, cT);
  gemm_bf16_wmma<EPI_BIAS_F32><<<dim3(cV / 128, (cB * cT) / 64, 1), 256, 0, stream>>>(
      ctxbf, headWT, 0, 0, cD, cD, cD, logits, nullptr, 0, cV, nullptr, 0, 0,
      head_b, 0.f);

  k_zero<<<1, 32, 0, stream>>>(out, 1);
  k_loss<<<cB * cT, 256, 0, stream>>>(logits, toks, out, cV, 1.0f / cB);
}
